// RSSM_31954556682345
// MI455X (gfx1250) — matrix operand order repaired and checked by reading, not compiled, not run on hardware
//
#include <hip/hip_runtime.h>
#include <hip/hip_bf16.h>

// ---------------- problem dims ----------------
#define T_STEPS 64
#define BATCH   512
#define XD      1024
#define HD      1024
#define ZD      256
#define AD      64

typedef __attribute__((ext_vector_type(16))) __bf16 v16bf;
typedef __attribute__((ext_vector_type(8)))  float  v8f;
typedef __attribute__((ext_vector_type(4)))  unsigned int v4u;

union Frag {
    v16bf v;
    __bf16 h[16];
    unsigned short s[16];
    v4u q[2];
};

__device__ __forceinline__ unsigned short f2bf_bits(float x) {
    union { __bf16 h; unsigned short s; } c;
    c.h = (__bf16)x;
    return c.s;
}
__device__ __forceinline__ float eluf(float x)      { return x > 0.f ? x : expm1f(x); }
__device__ __forceinline__ float softplusf(float x) { return x > 20.f ? x : log1pf(__expf(x)); }
__device__ __forceinline__ float sigmoidf(float x)  { return 1.f / (1.f + __expf(-x)); }

// A-fragment loader: 16x32 bf16 tile per CDNA5 WMMA A layout.
// lanes 0-15: row m0+lane, K = kb..kb+7 (h[0..7]) and kb+16..kb+23 (h[8..15])
// lanes 16-31: row m0+lane-16, K = kb+8..15 and kb+24..31
template<bool F32>
__device__ __forceinline__ void load_a(Frag& f, const void* src,
                                       int lda, int row, int kk, int lane) {
    int off = (lane & 16) ? 8 : 0;
    if (F32) {
        const float4* p4 = (const float4*)((const float*)src + (size_t)row * lda + kk + off);
        float4 a0 = p4[0], a1 = p4[1], a2 = p4[4], a3 = p4[5];
        f.h[0] = (__bf16)a0.x; f.h[1] = (__bf16)a0.y; f.h[2] = (__bf16)a0.z; f.h[3] = (__bf16)a0.w;
        f.h[4] = (__bf16)a1.x; f.h[5] = (__bf16)a1.y; f.h[6] = (__bf16)a1.z; f.h[7] = (__bf16)a1.w;
        f.h[8] = (__bf16)a2.x; f.h[9] = (__bf16)a2.y; f.h[10] = (__bf16)a2.z; f.h[11] = (__bf16)a2.w;
        f.h[12] = (__bf16)a3.x; f.h[13] = (__bf16)a3.y; f.h[14] = (__bf16)a3.z; f.h[15] = (__bf16)a3.w;
    } else {
        const unsigned short* p = (const unsigned short*)src + (size_t)row * lda + kk + off;
        f.q[0] = *(const v4u*)p;
        f.q[1] = *(const v4u*)(p + 16);
    }
}

// B-fragment loader: weights pre-packed as [kchunk][ntile][lane][16] bf16.
__device__ __forceinline__ void load_b(Frag& f, const unsigned short* Bp,
                                       int kc, int ntilesB, int nt, int lane) {
    const unsigned short* p = Bp + (((size_t)kc * ntilesB + nt) << 9) + (lane << 4);
    f.q[0] = *(const v4u*)p;
    f.q[1] = *(const v4u*)(p + 8);
}

__device__ __forceinline__ v8f wmma_bf16(const Frag& a, const Frag& b, v8f c) {
    return __builtin_amdgcn_wmma_f32_16x16x32_bf16(false, a.v, false, b.v,
                                                   (short)0, c, false, false);
}

// ---------------- prep kernels ----------------
__global__ __launch_bounds__(256) void pack_kernel(const float* __restrict__ W,
                                                   unsigned short* __restrict__ P,
                                                   int K, int N) {
    int idx = blockIdx.x * 256 + threadIdx.x;
    if (idx >= K * N) return;
    int i    = idx & 15;
    int lane = (idx >> 4) & 31;
    int frag = idx >> 9;
    int ntiles = N >> 4;
    int kc = frag / ntiles;
    int nt = frag - kc * ntiles;
    int ncol = (nt << 4) + (lane & 15);
    int k    = (kc << 5) + ((lane >> 4) << 3) + (i & 7) + ((i >> 3) << 4);
    P[idx] = f2bf_bits(W[(size_t)k * N + ncol]);
}

__global__ __launch_bounds__(256) void zero_kernel(float* __restrict__ b0,
                                                   unsigned short* __restrict__ h0) {
    int i = blockIdx.x * 256 + threadIdx.x;
    b0[i] = 0.f;
    h0[i] = 0;
}

// ---------------- generic GEMM: out_bf16 = elu(concat(A1,A2) @ Bp + bias) ----------------
// M=512 via grid; N multiple of 64; each wave computes a 16x64 strip (4 n-tiles).
// Double-buffered A/B fragments: chunk kc+1 is prefetched before chunk kc's WMMAs.
template<bool A1F32, bool HASA2, bool A2F32>
__global__ __launch_bounds__(256) void gemm_elu(
    const void* __restrict__ A1, int K1,
    const void* __restrict__ A2, int K2,
    const unsigned short* __restrict__ Bp, const float* __restrict__ bias,
    int N, unsigned short* __restrict__ outBf) {
    int lane = threadIdx.x & 31;
    int w = (blockIdx.x * blockDim.x + threadIdx.x) >> 5;
    int ntx = N >> 6;
    int mt = w / ntx, ng = w - mt * ntx;
    int m0 = mt << 4, n0 = ng << 6;
    int row = m0 + (lane & 15);
    int ntilesB = N >> 4;
    int nt0 = n0 >> 4;
    v8f acc0 = {}, acc1 = {}, acc2 = {}, acc3 = {};
    Frag a[2];
    Frag b[2][4];
    int nch1 = K1 >> 5;
    int ncht = nch1 + (HASA2 ? (K2 >> 5) : 0);
    load_a<A1F32>(a[0], A1, K1, row, 0, lane);
#pragma unroll
    for (int j = 0; j < 4; ++j) load_b(b[0][j], Bp, 0, ntilesB, nt0 + j, lane);
    for (int kc = 0; kc < ncht; ++kc) {
        int cur = kc & 1, nxt = cur ^ 1;
        int kn = kc + 1;
        if (kn < ncht) {
            if (!HASA2 || kn < nch1)
                load_a<A1F32>(a[nxt], A1, K1, row, kn << 5, lane);
            else
                load_a<A2F32>(a[nxt], A2, K2, row, (kn - nch1) << 5, lane);
#pragma unroll
            for (int j = 0; j < 4; ++j) load_b(b[nxt][j], Bp, kn, ntilesB, nt0 + j, lane);
        }
        acc0 = wmma_bf16(a[cur], b[cur][0], acc0);
        acc1 = wmma_bf16(a[cur], b[cur][1], acc1);
        acc2 = wmma_bf16(a[cur], b[cur][2], acc2);
        acc3 = wmma_bf16(a[cur], b[cur][3], acc3);
    }
    int rbase = m0 + ((lane >> 4) << 3);
    int colb = lane & 15;
#pragma unroll
    for (int j = 0; j < 4; ++j) {
        v8f acc = (j == 0) ? acc0 : (j == 1) ? acc1 : (j == 2) ? acc2 : acc3;
        int col = n0 + (j << 4) + colb;
        float bv = bias[col];
#pragma unroll
        for (int r = 0; r < 8; ++r) {
            float v = eluf(acc[r] + bv);
            outBf[(size_t)(rbase + r) * N + col] = f2bf_bits(v);
        }
    }
}

// ---------------- fused loc/scale (+rsample) heads; gridDim.z: 0=q, 1=p ----------------
__global__ __launch_bounds__(256) void locscale_kernel(
    const unsigned short* __restrict__ q_hid, const unsigned short* __restrict__ p_hid,
    const unsigned short* __restrict__ qWm, const unsigned short* __restrict__ qWs,
    const unsigned short* __restrict__ pWm, const unsigned short* __restrict__ pWs,
    const float* __restrict__ q_bm, const float* __restrict__ q_bs,
    const float* __restrict__ p_bm, const float* __restrict__ p_bs,
    const float* __restrict__ eps_t,
    float* __restrict__ qloc_t, float* __restrict__ qscale_t,
    float* __restrict__ ploc_t, float* __restrict__ pscale_t,
    float* __restrict__ zq_t, unsigned short* __restrict__ z_bf) {
    int lane = threadIdx.x & 31;
    int w = (blockIdx.x * blockDim.x + threadIdx.x) >> 5;
    int job = blockIdx.z;
    int mt = w >> 4, nt = w & 15;           // ZD/16 = 16 n-tiles
    int m0 = mt << 4, n0 = nt << 4;
    int row = m0 + (lane & 15);
    const unsigned short* A  = job ? p_hid : q_hid;
    const unsigned short* Bm = job ? pWm : qWm;
    const unsigned short* Bs = job ? pWs : qWs;
    v8f am = {}, as = {};
    const int ntilesB = ZD >> 4;            // 16
    const int NCH = HD >> 5;                // 32
    Frag a[2], bm2[2], bs2[2];
    load_a<false>(a[0], A, HD, row, 0, lane);
    load_b(bm2[0], Bm, 0, ntilesB, nt, lane);
    load_b(bs2[0], Bs, 0, ntilesB, nt, lane);
    for (int kc = 0; kc < NCH; ++kc) {
        int cur = kc & 1, nxt = cur ^ 1;
        if (kc + 1 < NCH) {
            load_a<false>(a[nxt], A, HD, row, (kc + 1) << 5, lane);
            load_b(bm2[nxt], Bm, kc + 1, ntilesB, nt, lane);
            load_b(bs2[nxt], Bs, kc + 1, ntilesB, nt, lane);
        }
        am = wmma_bf16(a[cur], bm2[cur], am);
        as = wmma_bf16(a[cur], bs2[cur], as);
    }
    int rbase = m0 + ((lane >> 4) << 3);
    int col = n0 + (lane & 15);
    float bm = (job ? p_bm : q_bm)[col];
    float bs = (job ? p_bs : q_bs)[col];
#pragma unroll
    for (int r = 0; r < 8; ++r) {
        size_t o = (size_t)(rbase + r) * ZD + col;
        float loc = am[r] + bm;
        float scl = softplusf(as[r] + bs);
        if (job == 0) {
            qloc_t[o] = loc;
            qscale_t[o] = scl;
            float z = loc + scl * eps_t[o];
            zq_t[o] = z;
            z_bf[o] = f2bf_bits(z);
        } else {
            ploc_t[o] = loc;
            pscale_t[o] = scl;
        }
    }
}

// ---------------- fully fused GRU cell: 6 WMMA chains per 16x16 h_new tile ----------------
__global__ __launch_bounds__(256) void gru_kernel(
    const unsigned short* __restrict__ a_bf, const unsigned short* __restrict__ h_bf,
    const unsigned short* __restrict__ Wih, const unsigned short* __restrict__ Whh,
    const float* __restrict__ bih, const float* __restrict__ bhh,
    const float* __restrict__ h_old, float* __restrict__ h_new,
    unsigned short* __restrict__ hbf_new) {
    int lane = threadIdx.x & 31;
    int w = (blockIdx.x * blockDim.x + threadIdx.x) >> 5;
    int mt = w >> 6, nt = w & 63;           // HD/16 = 64 n-tiles
    int m0 = mt << 4, n0 = nt << 4;
    int row = m0 + (lane & 15);
    v8f aI0 = {}, aI1 = {}, aI2 = {}, aH0 = {}, aH1 = {}, aH2 = {};
    const int ntB = (3 * HD) >> 4;          // 192
    const int NCH = HD >> 5;                // 32
    Frag fa[2], fh[2];
    load_a<false>(fa[0], a_bf, HD, row, 0, lane);
    load_a<false>(fh[0], h_bf, HD, row, 0, lane);
    for (int kc = 0; kc < NCH; ++kc) {
        int cur = kc & 1, nxt = cur ^ 1;
        Frag b0, b1, b2, b3, b4, b5;
        load_b(b0, Wih, kc, ntB, nt,       lane);
        load_b(b1, Wih, kc, ntB, nt + 64,  lane);
        load_b(b2, Wih, kc, ntB, nt + 128, lane);
        load_b(b3, Whh, kc, ntB, nt,       lane);
        load_b(b4, Whh, kc, ntB, nt + 64,  lane);
        load_b(b5, Whh, kc, ntB, nt + 128, lane);
        if (kc + 1 < NCH) {
            load_a<false>(fa[nxt], a_bf, HD, row, (kc + 1) << 5, lane);
            load_a<false>(fh[nxt], h_bf, HD, row, (kc + 1) << 5, lane);
        }
        aI0 = wmma_bf16(fa[cur], b0, aI0);
        aI1 = wmma_bf16(fa[cur], b1, aI1);
        aI2 = wmma_bf16(fa[cur], b2, aI2);
        aH0 = wmma_bf16(fh[cur], b3, aH0);
        aH1 = wmma_bf16(fh[cur], b4, aH1);
        aH2 = wmma_bf16(fh[cur], b5, aH2);
    }
    int rbase = m0 + ((lane >> 4) << 3);
    int col = n0 + (lane & 15);
    float bi_r = bih[col], bi_z = bih[col + HD], bi_n = bih[col + 2 * HD];
    float bh_r = bhh[col], bh_z = bhh[col + HD], bh_n = bhh[col + 2 * HD];
#pragma unroll
    for (int r = 0; r < 8; ++r) {
        size_t o = (size_t)(rbase + r) * HD + col;
        float gr = sigmoidf(aI0[r] + bi_r + aH0[r] + bh_r);
        float gz = sigmoidf(aI1[r] + bi_z + aH1[r] + bh_z);
        float gn = tanhf(aI2[r] + bi_n + gr * (aH2[r] + bh_n));
        float hv = (1.f - gz) * gn + gz * h_old[o];
        h_new[o] = hv;
        hbf_new[o] = f2bf_bits(hv);
    }
}

// ---------------- host ----------------
extern "C" void kernel_launch(void* const* d_in, const int* in_sizes, int n_in,
                              void* d_out, int out_size, void* d_ws, size_t ws_size,
                              hipStream_t stream) {
    (void)in_sizes; (void)n_in; (void)out_size; (void)ws_size;
    const float* obs   = (const float*)d_in[0];
    const float* act   = (const float*)d_in[1];
    const float* eps   = (const float*)d_in[2];
    const float* aggrW = (const float*)d_in[3];
    const float* aggrb = (const float*)d_in[4];
    const float* gWih  = (const float*)d_in[5];
    const float* gWhh  = (const float*)d_in[6];
    const float* gbih  = (const float*)d_in[7];
    const float* gbhh  = (const float*)d_in[8];
    const float* qW1   = (const float*)d_in[9];
    const float* qb1   = (const float*)d_in[10];
    const float* qWm   = (const float*)d_in[11];
    const float* qbm   = (const float*)d_in[12];
    const float* qWs   = (const float*)d_in[13];
    const float* qbs   = (const float*)d_in[14];
    const float* pW1   = (const float*)d_in[15];
    const float* pb1   = (const float*)d_in[16];
    const float* pWm   = (const float*)d_in[17];
    const float* pbm   = (const float*)d_in[18];
    const float* pWs   = (const float*)d_in[19];
    const float* pbs   = (const float*)d_in[20];

    unsigned short* ws = (unsigned short*)d_ws;
    size_t off = 0;
    auto alloc = [&](size_t n) { unsigned short* p = ws + off; off += n; return p; };
    unsigned short* qW1p  = alloc((size_t)2048 * 1024);
    unsigned short* pW1p  = alloc((size_t)1024 * 1024);
    unsigned short* qWmp  = alloc((size_t)1024 * 256);
    unsigned short* qWsp  = alloc((size_t)1024 * 256);
    unsigned short* pWmp  = alloc((size_t)1024 * 256);
    unsigned short* pWsp  = alloc((size_t)1024 * 256);
    unsigned short* aggrp = alloc((size_t)320 * 1024);
    unsigned short* Wihp  = alloc((size_t)1024 * 3072);
    unsigned short* Whhp  = alloc((size_t)1024 * 3072);
    unsigned short* qhid  = alloc((size_t)BATCH * HD);
    unsigned short* phid  = alloc((size_t)BATCH * HD);
    unsigned short* zbf   = alloc((size_t)BATCH * ZD);
    unsigned short* abf   = alloc((size_t)BATCH * HD);
    unsigned short* hbfA  = alloc((size_t)BATCH * HD);
    unsigned short* hbfB  = alloc((size_t)BATCH * HD);

    float* out = (float*)d_out;
    float* beliefs = out;                                   // [T+1, B, H]
    size_t SZ = (size_t)T_STEPS * BATCH * ZD;
    float* qloc   = out + (size_t)(T_STEPS + 1) * BATCH * HD;
    float* qscale = qloc + SZ;
    float* ploc   = qscale + SZ;
    float* pscale = ploc + SZ;
    float* zq     = pscale + SZ;

    auto pack = [&](const float* W, unsigned short* P, int K, int N) {
        pack_kernel<<<(K * N) / 256, 256, 0, stream>>>(W, P, K, N);
    };
    pack(qW1, qW1p, 2048, 1024);
    pack(pW1, pW1p, 1024, 1024);
    pack(qWm, qWmp, 1024, 256);
    pack(qWs, qWsp, 1024, 256);
    pack(pWm, pWmp, 1024, 256);
    pack(pWs, pWsp, 1024, 256);
    pack(aggrW, aggrp, 320, 1024);
    pack(gWih, Wihp, 1024, 3072);
    pack(gWhh, Whhp, 1024, 3072);

    // h0 = 0 (both f32 beliefs[0] and bf16 buffer)
    zero_kernel<<<(BATCH * HD) / 256, 256, 0, stream>>>(beliefs, hbfA);

    unsigned short* hb[2] = { hbfA, hbfB };

    for (int t = 0; t < T_STEPS; ++t) {
        const float* obs_t = obs + (size_t)t * BATCH * XD;
        const float* act_t = act + (size_t)t * BATCH * AD;
        const float* eps_t = eps + (size_t)t * BATCH * ZD;
        unsigned short* hcur  = hb[t & 1];
        unsigned short* hnext = hb[(t + 1) & 1];
        float* h_old = beliefs + (size_t)t * BATCH * HD;
        float* h_new = beliefs + (size_t)(t + 1) * BATCH * HD;
        size_t zo = (size_t)t * BATCH * ZD;

        // q_hid = elu([obs_t, h] @ q_W1 + q_b1)   (K = 1024+1024)
        gemm_elu<true, true, false><<<64, 256, 0, stream>>>(
            obs_t, XD, hcur, HD, qW1p, qb1, HD, qhid);
        // p_hid = elu(h @ p_W1 + p_b1)
        gemm_elu<false, false, false><<<64, 256, 0, stream>>>(
            hcur, HD, nullptr, 0, pW1p, pb1, HD, phid);
        // q/p loc+scale heads, rsample z_q
        dim3 gls(64, 1, 2);
        locscale_kernel<<<gls, 256, 0, stream>>>(qhid, phid, qWmp, qWsp, pWmp, pWsp,
                                                 qbm, qbs, pbm, pbs, eps_t,
                                                 qloc + zo, qscale + zo,
                                                 ploc + zo, pscale + zo,
                                                 zq + zo, zbf);
        // a = elu([z_q, act_t] @ aggr_W + aggr_b)   (K = 256+64)
        gemm_elu<false, true, true><<<64, 256, 0, stream>>>(
            zbf, ZD, act_t, AD, aggrp, aggrb, HD, abf);
        // GRU: h_new = (1-z)*n + z*h
        gru_kernel<<<256, 256, 0, stream>>>(abf, hcur, Wihp, Whhp, gbih, gbhh,
                                            h_old, h_new, hnext);
    }
}
